// EncoderLayer_936302871322
// MI455X (gfx1250) — matrix operand order repaired and checked by reading, not compiled, not run on hardware
//
#include <hip/hip_runtime.h>
#include <hip/hip_bf16.h>
#include <math.h>

// ---------------------------------------------------------------------------
// Types / helpers
// ---------------------------------------------------------------------------
typedef __attribute__((ext_vector_type(16))) __bf16 v16bf;
typedef __attribute__((ext_vector_type(8)))  float  v8f;

union Frag16 { v16bf v; unsigned u[8]; };
union AccF   { v8f v; float f[8]; };

__device__ __forceinline__ unsigned short f2bf(float f) {
  unsigned u = __builtin_bit_cast(unsigned, f);
  u += 0x7FFFu + ((u >> 16) & 1u);            // round-to-nearest-even
  return (unsigned short)(u >> 16);
}

// Low 32 bits of a flat pointer to __shared__ == LDS byte offset
// (aperture occupies addr[63:32] on gfx1250).
__device__ __forceinline__ unsigned lds_of(const void* p) {
  return (unsigned)(unsigned long long)p;
}

// CDNA5 async memory->LDS copy, 16B per lane, tracked by ASYNCcnt.
__device__ __forceinline__ void async_load_b128(unsigned lds_byte_off, const void* gaddr) {
  asm volatile("global_load_async_to_lds_b128 %0, %1, off"
               :: "v"(lds_byte_off), "v"(gaddr)
               : "memory");
}
__device__ __forceinline__ void wait_async0() {
  asm volatile("s_wait_asynccnt 0" ::: "memory");
}

// ---------------------------------------------------------------------------
// f32 -> bf16 conversion (grid-stride)
// ---------------------------------------------------------------------------
__global__ void convert_f32_bf16(const float* __restrict__ src,
                                 unsigned short* __restrict__ dst, int nelem) {
  int i = blockIdx.x * blockDim.x + threadIdx.x;
  const int stride = gridDim.x * blockDim.x;
  for (; i < nelem; i += stride) dst[i] = f2bf(src[i]);
}

// ---------------------------------------------------------------------------
// bf16 GEMM: C = A[MxK] * B[KxN] (+bias) (+epilogue), fp32 accumulation.
// Double-buffered LDS; A tiles via global_load_async_to_lds_b128.
// mode: 0 = f32 out (+optional residual)
//       1 = bf16 out with exact GELU
//       2 = bf16 out
//       3 = bf16 out scattered to [n, h, l, dk] head layout (L=512, dk=64)
// ---------------------------------------------------------------------------
#define TB_M 128
#define TB_N 128
#define TB_K 32
#define LDA_SH 40   // padded A row (elements): 16B-aligned vectors, contiguous frag halves
#define LDB_SH 34   // padded Bt row (elements): 4B-aligned packed pair loads

__global__ __launch_bounds__(256)
void gemm_bf16_kernel(const unsigned short* __restrict__ A,
                      const unsigned short* __restrict__ B,
                      const float* __restrict__ bias,
                      const float* __restrict__ resid,
                      void* __restrict__ Out,
                      int M, int N, int K, int mode) {
  __shared__ unsigned short As[2][TB_M * LDA_SH];
  __shared__ unsigned short Bs[2][TB_N * LDB_SH];

  const int tid  = threadIdx.x;
  const int lane = tid & 31;
  const int wave = tid >> 5;
  const int cl   = lane & 15;
  const int hi   = lane >> 4;
  const int wm   = wave & 1;     // row block of 64
  const int wn   = wave >> 1;    // col block of 32

  const int m0 = blockIdx.y * TB_M;
  const int n0 = blockIdx.x * TB_N;

  v8f zero = {0.f, 0.f, 0.f, 0.f, 0.f, 0.f, 0.f, 0.f};
  v8f acc[4][2];
#pragma unroll
  for (int a = 0; a < 4; ++a)
#pragma unroll
    for (int b = 0; b < 2; ++b) acc[a][b] = zero;

  const int r_a = tid >> 1;   // A tile row (0..127), 2 threads/row
  const int ha  = tid & 1;    // which 16-element half of K
  const int k_b = tid >> 3;   // B tile k (0..31), 8 threads/row
  const int nb8 = tid & 7;    // 16-col chunk along N

  const unsigned short* Ag = A + (size_t)(m0 + r_a) * K + ha * 16;
  const unsigned short* Bg = B + (size_t)k_b * N + n0 + nb8 * 16;

  const int T = K / TB_K;

  // ---- prologue: tile 0 into buffer 0 ----
  {
    const unsigned d0 = lds_of(&As[0][r_a * LDA_SH + ha * 16]);
    async_load_b128(d0,      Ag);
    async_load_b128(d0 + 16, Ag + 8);
    const uint4* bsrc = (const uint4*)Bg;
    uint4 t0 = bsrc[0], t1 = bsrc[1];
    union { uint4 q[2]; unsigned short s[16]; } u;
    u.q[0] = t0; u.q[1] = t1;
#pragma unroll
    for (int j = 0; j < 16; ++j)
      Bs[0][(nb8 * 16 + j) * LDB_SH + k_b] = u.s[j];
    wait_async0();
  }
  __syncthreads();

  for (int t = 0; t < T; ++t) {
    const int cur = t & 1;
    const int nxt = cur ^ 1;
    const bool havenext = (t + 1 < T);

    // ---- issue next tile's copies (async A; B staged in VGPRs) ----
    uint4 bt0, bt1;
    if (havenext) {
      const int k1 = (t + 1) * TB_K;
      const unsigned short* as = Ag + k1;
      const unsigned d0 = lds_of(&As[nxt][r_a * LDA_SH + ha * 16]);
      async_load_b128(d0,      as);
      async_load_b128(d0 + 16, as + 8);
      const uint4* bsrc = (const uint4*)(Bg + (size_t)k1 * N);
      bt0 = bsrc[0]; bt1 = bsrc[1];
    }

    // ---- compute tile t from buffer cur ----
    Frag16 af[4];
#pragma unroll
    for (int tm = 0; tm < 4; ++tm) {
      const int r = (wm * 64 + tm * 16 + cl) * LDA_SH;
#pragma unroll
      for (int vv = 0; vv < 8; ++vv) {
        const int kd = (hi << 4) + (vv << 1);
        af[tm].u[vv] = *(const unsigned*)&As[cur][r + kd];
      }
    }
    Frag16 bfr[2];
#pragma unroll
    for (int tn = 0; tn < 2; ++tn) {
      const int c = (wn * 32 + tn * 16 + cl) * LDB_SH;
#pragma unroll
      for (int vv = 0; vv < 8; ++vv)
        bfr[tn].u[vv] = *(const unsigned*)&Bs[cur][c + (hi << 4) + (vv << 1)];
    }
#pragma unroll
    for (int tm = 0; tm < 4; ++tm)
#pragma unroll
      for (int tn = 0; tn < 2; ++tn)
        acc[tm][tn] = __builtin_amdgcn_wmma_f32_16x16x32_bf16(
            false, af[tm].v, false, bfr[tn].v, (short)0, acc[tm][tn], false, false);

    // ---- finish next tile's B transpose-store + async drain ----
    if (havenext) {
      union { uint4 q[2]; unsigned short s[16]; } u;
      u.q[0] = bt0; u.q[1] = bt1;
#pragma unroll
      for (int j = 0; j < 16; ++j)
        Bs[nxt][(nb8 * 16 + j) * LDB_SH + k_b] = u.s[j];
      wait_async0();
    }
    __syncthreads();
  }

  // --- epilogue ---
#pragma unroll
  for (int tm = 0; tm < 4; ++tm) {
#pragma unroll
    for (int tn = 0; tn < 2; ++tn) {
      const int gc = n0 + wn * 32 + tn * 16 + cl;
      const float bv = bias ? bias[gc] : 0.0f;
      AccF t; t.v = acc[tm][tn];
#pragma unroll
      for (int vv = 0; vv < 8; ++vv) {
        const int gr = m0 + wm * 64 + tm * 16 + vv + (hi << 3);
        float val = t.f[vv] + bv;
        if (mode == 0) {
          if (resid) val += resid[(size_t)gr * N + gc];
          ((float*)Out)[(size_t)gr * N + gc] = val;
        } else if (mode == 1) {
          val = 0.5f * val * (1.0f + erff(val * 0.70710678118654752f));
          ((unsigned short*)Out)[(size_t)gr * N + gc] = f2bf(val);
        } else if (mode == 2) {
          ((unsigned short*)Out)[(size_t)gr * N + gc] = f2bf(val);
        } else {                     // QKV scatter -> [n, h, l, dk]
          const int li = gr & 511, ni = gr >> 9;
          const int hh = gc >> 6,  dd = gc & 63;
          ((unsigned short*)Out)[(((size_t)(ni * 16 + hh) * 512 + li) << 6) + dd] = f2bf(val);
        }
      }
    }
  }
}

// ---------------------------------------------------------------------------
// Flash attention: one block per (64 q-rows, head, batch); 4 waves, one 16-row
// q tile per wave; online softmax; ctx written bf16 to [n, l, h*64+dk].
// K tile staged via async-to-LDS.
// ---------------------------------------------------------------------------
__global__ __launch_bounds__(128)
void attn_kernel(const unsigned short* __restrict__ q,
                 const unsigned short* __restrict__ k,
                 const unsigned short* __restrict__ v,
                 const unsigned char* __restrict__ mask,
                 unsigned short* __restrict__ ctx) {
  __shared__ unsigned short Ks[32 * 64];        // [key][dk]
  __shared__ unsigned short Vt[64 * 32];        // [dk][key]
  __shared__ unsigned short Ps[4][16 * 36];     // per-wave P scratch

  const int tid  = threadIdx.x;
  const int lane = tid & 31;
  const int wave = tid >> 5;
  const int cl   = lane & 15, hi = lane >> 4;
  const int n = blockIdx.z, h = blockIdx.y;
  const int q0 = blockIdx.x * 64 + wave * 16;
  const size_t headoff = ((size_t)(n * 16 + h)) * 512 * 64;

  // Q A-fragments (dk 0..31 and 32..63)
  Frag16 aq0, aq1;
  {
    const unsigned short* qp = q + headoff + (size_t)(q0 + cl) * 64;
#pragma unroll
    for (int vv = 0; vv < 8; ++vv) {
      const int kd = (hi << 4) + (vv << 1);
      aq0.u[vv] = *(const unsigned*)&qp[kd];
      aq1.u[vv] = *(const unsigned*)&qp[kd + 32];
    }
  }

  float m_i[8], l_i[8];
#pragma unroll
  for (int vv = 0; vv < 8; ++vv) { m_i[vv] = -INFINITY; l_i[vv] = 0.0f; }
  v8f zero = {0.f, 0.f, 0.f, 0.f, 0.f, 0.f, 0.f, 0.f};
  v8f accv[4];
#pragma unroll
  for (int dt = 0; dt < 4; ++dt) accv[dt] = zero;

  const int kl_key = tid >> 2;          // 0..31
  const int kl_c   = (tid & 3) * 16;    // dk chunk

  for (int kb = 0; kb < 16; ++kb) {
    const int key0 = kb * 32;
    { // K tile (row-major) via async-to-LDS
      const unsigned short* ks = k + headoff + (size_t)(key0 + kl_key) * 64 + kl_c;
      const unsigned d = lds_of(&Ks[kl_key * 64 + kl_c]);
      async_load_b128(d,      ks);
      async_load_b128(d + 16, ks + 8);
    }
    { // V tile transposed (needs VGPR pass)
      const unsigned short* vp = v + headoff + (size_t)(key0 + kl_key) * 64 + kl_c;
#pragma unroll
      for (int j = 0; j < 8; ++j) {
        unsigned pr = *(const unsigned*)&vp[2 * j];
        Vt[(kl_c + 2 * j)     * 32 + kl_key] = (unsigned short)(pr & 0xFFFFu);
        Vt[(kl_c + 2 * j + 1) * 32 + kl_key] = (unsigned short)(pr >> 16);
      }
    }
    wait_async0();
    __syncthreads();

    // scores 16x32 = two 16x16 tiles, each 2 WMMAs over dk=64
    AccF s[2];
#pragma unroll
    for (int st = 0; st < 2; ++st) {
      Frag16 b0, b1;
      const int rb = (st * 16 + cl) * 64;
#pragma unroll
      for (int vv = 0; vv < 8; ++vv) {
        const int kd = (hi << 4) + (vv << 1);
        b0.u[vv] = *(const unsigned*)&Ks[rb + kd];
        b1.u[vv] = *(const unsigned*)&Ks[rb + kd + 32];
      }
      v8f sv = __builtin_amdgcn_wmma_f32_16x16x32_bf16(false, aq0.v, false, b0.v, (short)0, zero, false, false);
      sv = __builtin_amdgcn_wmma_f32_16x16x32_bf16(false, aq1.v, false, b1.v, (short)0, sv, false, false);
      s[st].v = sv;
#pragma unroll
      for (int vv = 0; vv < 8; ++vv) {
        float sc = s[st].f[vv] * 0.125f;   // 1/sqrt(64)
        const int qr = q0 + vv + (hi << 3);
        const int kc = key0 + st * 16 + cl;
        if (mask[((size_t)n * 512 + qr) * 512 + kc]) sc = -INFINITY;
        s[st].f[vv] = sc;
      }
    }

    // online softmax (row reductions across 16-lane halves)
    float cfa[8];
#pragma unroll
    for (int vv = 0; vv < 8; ++vv) {
      float r = fmaxf(s[0].f[vv], s[1].f[vv]);
      r = fmaxf(r, __shfl_xor(r, 1, 32));
      r = fmaxf(r, __shfl_xor(r, 2, 32));
      r = fmaxf(r, __shfl_xor(r, 4, 32));
      r = fmaxf(r, __shfl_xor(r, 8, 32));
      const float mn = fmaxf(m_i[vv], r);
      const float cf = (mn == -INFINITY) ? 1.0f : expf(m_i[vv] - mn);
      const float p0 = (mn == -INFINITY) ? 0.0f : expf(s[0].f[vv] - mn);
      const float p1 = (mn == -INFINITY) ? 0.0f : expf(s[1].f[vv] - mn);
      s[0].f[vv] = p0; s[1].f[vv] = p1;
      float rs = p0 + p1;
      rs += __shfl_xor(rs, 1, 32);
      rs += __shfl_xor(rs, 2, 32);
      rs += __shfl_xor(rs, 4, 32);
      rs += __shfl_xor(rs, 8, 32);
      l_i[vv] = l_i[vv] * cf + rs;
      m_i[vv] = mn;
      cfa[vv] = cf;
    }
#pragma unroll
    for (int dt = 0; dt < 4; ++dt) {
      AccF a; a.v = accv[dt];
#pragma unroll
      for (int vv = 0; vv < 8; ++vv) a.f[vv] *= cfa[vv];
      accv[dt] = a.v;
    }

    // P (C-layout) -> per-wave LDS -> A-fragment layout
#pragma unroll
    for (int st = 0; st < 2; ++st)
#pragma unroll
      for (int vv = 0; vv < 8; ++vv)
        Ps[wave][(vv + (hi << 3)) * 36 + st * 16 + cl] = f2bf(s[st].f[vv]);

    Frag16 ap;
#pragma unroll
    for (int vv = 0; vv < 8; ++vv) {
      const int kd = (hi << 4) + (vv << 1);
      ap.u[vv] = *(const unsigned*)&Ps[wave][cl * 36 + kd];
    }
#pragma unroll
    for (int dt = 0; dt < 4; ++dt) {
      Frag16 bv;
      const int rb = (dt * 16 + cl) * 32;
#pragma unroll
      for (int vv = 0; vv < 8; ++vv)
        bv.u[vv] = *(const unsigned*)&Vt[rb + (hi << 4) + (vv << 1)];
      accv[dt] = __builtin_amdgcn_wmma_f32_16x16x32_bf16(false, ap.v, false, bv.v, (short)0, accv[dt], false, false);
    }
    __syncthreads();
  }

  float inv[8];
#pragma unroll
  for (int vv = 0; vv < 8; ++vv) inv[vv] = (l_i[vv] > 0.0f) ? (1.0f / l_i[vv]) : 0.0f;
#pragma unroll
  for (int dt = 0; dt < 4; ++dt) {
    AccF a; a.v = accv[dt];
#pragma unroll
    for (int vv = 0; vv < 8; ++vv) {
      const int row = q0 + vv + (hi << 3);
      const int col = h * 64 + dt * 16 + cl;
      ctx[((size_t)(n * 512 + row)) * 1024 + col] = f2bf(a.f[vv] * inv[vv]);
    }
  }
}

// ---------------------------------------------------------------------------
// LayerNorm over D=1024: one block (256 threads) per row; optional bf16 copy.
// ---------------------------------------------------------------------------
__global__ __launch_bounds__(256)
void layernorm_kernel(const float* __restrict__ x, const float* __restrict__ g,
                      const float* __restrict__ b, float* __restrict__ out,
                      unsigned short* __restrict__ out_bf) {
  __shared__ float red_s[8], red_q[8];
  const int row = blockIdx.x;
  const int tid = threadIdx.x;
  const float4 xv = ((const float4*)(x + (size_t)row * 1024))[tid];
  float s  = xv.x + xv.y + xv.z + xv.w;
  float qq = xv.x * xv.x + xv.y * xv.y + xv.z * xv.z + xv.w * xv.w;
#pragma unroll
  for (int off = 16; off > 0; off >>= 1) {
    s  += __shfl_xor(s,  off, 32);
    qq += __shfl_xor(qq, off, 32);
  }
  if ((tid & 31) == 0) { red_s[tid >> 5] = s; red_q[tid >> 5] = qq; }
  __syncthreads();
  float ts = 0.0f, tq = 0.0f;
#pragma unroll
  for (int i = 0; i < 8; ++i) { ts += red_s[i]; tq += red_q[i]; }
  const float mu   = ts * (1.0f / 1024.0f);
  const float var  = tq * (1.0f / 1024.0f) - mu * mu;
  const float rstd = rsqrtf(var + 1e-5f);
  const float4 gv = ((const float4*)g)[tid];
  const float4 bv = ((const float4*)b)[tid];
  float4 ov;
  ov.x = (xv.x - mu) * rstd * gv.x + bv.x;
  ov.y = (xv.y - mu) * rstd * gv.y + bv.y;
  ov.z = (xv.z - mu) * rstd * gv.z + bv.z;
  ov.w = (xv.w - mu) * rstd * gv.w + bv.w;
  ((float4*)(out + (size_t)row * 1024))[tid] = ov;
  if (out_bf) {
    unsigned short* ob = out_bf + (size_t)row * 1024 + tid * 4;
    ob[0] = f2bf(ov.x); ob[1] = f2bf(ov.y); ob[2] = f2bf(ov.z); ob[3] = f2bf(ov.w);
  }
}

// ---------------------------------------------------------------------------
// Host-side orchestration
// ---------------------------------------------------------------------------
extern "C" void kernel_launch(void* const* d_in, const int* in_sizes, int n_in,
                              void* d_out, int out_size, void* d_ws, size_t ws_size,
                              hipStream_t stream) {
  (void)in_sizes; (void)n_in; (void)out_size; (void)ws_size;
  const float* x    = (const float*)d_in[0];
  const unsigned char* mask = (const unsigned char*)d_in[1];
  const float* WQ = (const float*)d_in[2];  const float* bQ  = (const float*)d_in[3];
  const float* WK = (const float*)d_in[4];  const float* bK  = (const float*)d_in[5];
  const float* WV = (const float*)d_in[6];  const float* bV  = (const float*)d_in[7];
  const float* WO = (const float*)d_in[8];  const float* bO  = (const float*)d_in[9];
  const float* ln0g = (const float*)d_in[10]; const float* ln0b = (const float*)d_in[11];
  const float* W1 = (const float*)d_in[12]; const float* b1  = (const float*)d_in[13];
  const float* W2 = (const float*)d_in[14]; const float* b2  = (const float*)d_in[15];
  const float* ln1g = (const float*)d_in[16]; const float* ln1b = (const float*)d_in[17];

  const int D = 1024, FF = 4096, NB = 16, LS = 512;
  const int M = NB * LS;   // 8192

  char* base = (char*)d_ws;
  size_t off = 0;
  auto take = [&](size_t bytes) -> char* {
    char* p = base + off;
    off += (bytes + 255) & ~(size_t)255;
    return p;
  };
  unsigned short* wqb = (unsigned short*)take((size_t)D * D * 2);
  unsigned short* wkb = (unsigned short*)take((size_t)D * D * 2);
  unsigned short* wvb = (unsigned short*)take((size_t)D * D * 2);
  unsigned short* wob = (unsigned short*)take((size_t)D * D * 2);
  unsigned short* w1b = (unsigned short*)take((size_t)D * FF * 2);
  unsigned short* w2b = (unsigned short*)take((size_t)FF * D * 2);
  unsigned short* xb  = (unsigned short*)take((size_t)M * D * 2);
  unsigned short* qb  = (unsigned short*)take((size_t)M * D * 2);
  unsigned short* kb  = (unsigned short*)take((size_t)M * D * 2);
  unsigned short* vb  = (unsigned short*)take((size_t)M * D * 2);
  float*          y0  = (float*)take((size_t)M * D * 4);
  float*          hbuf= (float*)take((size_t)M * D * 4);
  unsigned short* gelub = (unsigned short*)take((size_t)M * FF * 2);
  // aliases (dead buffers reused)
  unsigned short* ctxb = xb;   // x bf16 dead after QKV projections
  unsigned short* hb16 = qb;   // q dead after attention
  float*          y1   = y0;   // y0 dead after LN0

  // 1) fp32 -> bf16 conversions
  convert_f32_bf16<<<2048, 256, 0, stream>>>(WQ, wqb, D * D);
  convert_f32_bf16<<<2048, 256, 0, stream>>>(WK, wkb, D * D);
  convert_f32_bf16<<<2048, 256, 0, stream>>>(WV, wvb, D * D);
  convert_f32_bf16<<<2048, 256, 0, stream>>>(WO, wob, D * D);
  convert_f32_bf16<<<2048, 256, 0, stream>>>(W1, w1b, D * FF);
  convert_f32_bf16<<<2048, 256, 0, stream>>>(W2, w2b, FF * D);
  convert_f32_bf16<<<2048, 256, 0, stream>>>(x,  xb,  M * D);

  // 2) QKV projections (scatter to [n,h,l,dk] bf16)
  dim3 gq(D / 128, M / 128);
  gemm_bf16_kernel<<<gq, 256, 0, stream>>>(xb, wqb, bQ, nullptr, qb, M, D, D, 3);
  gemm_bf16_kernel<<<gq, 256, 0, stream>>>(xb, wkb, bK, nullptr, kb, M, D, D, 3);
  gemm_bf16_kernel<<<gq, 256, 0, stream>>>(xb, wvb, bV, nullptr, vb, M, D, D, 3);

  // 3) attention -> ctx bf16 [n,l,h*64+dk]
  attn_kernel<<<dim3(LS / 64, 16, NB), 128, 0, stream>>>(qb, kb, vb, mask, ctxb);

  // 4) output projection + residual(x) -> y0 f32
  gemm_bf16_kernel<<<gq, 256, 0, stream>>>(ctxb, wob, bO, x, y0, M, D, D, 0);

  // 5) LayerNorm0 -> h f32 + h bf16
  layernorm_kernel<<<M, 256, 0, stream>>>(y0, ln0g, ln0b, hbuf, hb16);

  // 6) FFN1 + GELU -> bf16
  dim3 gf1(FF / 128, M / 128);
  gemm_bf16_kernel<<<gf1, 256, 0, stream>>>(hb16, w1b, b1, nullptr, gelub, M, FF, D, 1);

  // 7) FFN2 + residual(h) -> y1 f32
  gemm_bf16_kernel<<<gq, 256, 0, stream>>>(gelub, w2b, b2, hbuf, y1, M, D, FF, 0);

  // 8) LayerNorm1 -> d_out f32
  layernorm_kernel<<<M, 256, 0, stream>>>(y1, ln1g, ln1b, (float*)d_out, nullptr);
}
